// Processor_31473520345827
// MI455X (gfx1250) — matrix operand order repaired and checked by reading, not compiled, not run on hardware
//
#include <hip/hip_runtime.h>

// ---------------- problem constants ----------------
#define N_NODES  40962
#define N_EDGES  245772
#define LATENT   256
#define HIDDEN   256
#define HEADS    4
#define NUM_FREQ 32

typedef unsigned short u16;
typedef __attribute__((ext_vector_type(16))) __bf16 v16bf;
typedef __attribute__((ext_vector_type(8)))  __bf16 v8bf;
typedef __attribute__((ext_vector_type(8)))  float  v8f;

#define DEV static __device__ __forceinline__

DEV u16 f2bf(float f) {                       // round-to-nearest-even f32 -> bf16 bits
  unsigned u = __float_as_uint(f);
  u += 0x7FFFu + ((u >> 16) & 1u);
  return (u16)(u >> 16);
}
DEV float bf2f(u16 h) { return __uint_as_float(((unsigned)h) << 16); }

DEV float wred32(float v) {                   // full wave32 xor reduction (all lanes get sum)
  v += __shfl_xor(v, 1, 32);  v += __shfl_xor(v, 2, 32);
  v += __shfl_xor(v, 4, 32);  v += __shfl_xor(v, 8, 32);
  v += __shfl_xor(v, 16, 32);
  return v;
}
DEV float red8(float v) {                     // reduce within groups of 8 lanes
  v += __shfl_xor(v, 1, 32); v += __shfl_xor(v, 2, 32); v += __shfl_xor(v, 4, 32);
  return v;
}
DEV float atomicMaxF(float* a, float v) {     // IEEE-order float max via int punning
  if (v >= 0.f) return __int_as_float(atomicMax((int*)a, __float_as_int(v)));
  return __uint_as_float(atomicMin((unsigned*)a, __float_as_uint(v)));
}

// ---------------- bf16 WMMA GEMM: C[M,N] = A[M,K] * BT[N,K]^T + bias ----------------
// One wave per 16x64 output tile: one A fragment feeds 4 v_wmma_f32_16x16x32_bf16
// against 4 B column-groups (fixed j*16*K offsets -> fits the 24-bit inst offset).
__global__ void gemm_bf16_kernel(const u16* __restrict__ A, const u16* __restrict__ BT,
                                 const float* __restrict__ bias, void* __restrict__ C,
                                 int out_bf16, long M, int N, int K)
{
  const int lane = threadIdx.x & 31;
  const int wave = threadIdx.x >> 5;
  const long tiles_n = (long)(N >> 6);            // 64-wide column tiles
  const long tiles_m = (M + 15) >> 4;
  const long tile = (long)blockIdx.x * (blockDim.x >> 5) + wave;
  if (tile >= tiles_m * tiles_n) return;
  const long tm  = tile / tiles_n;
  const int  tn4 = (int)(tile - tm * tiles_n);

  const int i = lane & 15;          // A row / B col within tile (ISA lane layout)
  const int g = lane >> 4;          // lane half selects K sub-block
  long arow = tm * 16 + i; if (arow >= M) arow = M - 1;   // clamp ragged tail (stores guarded)
  const u16* ap = A  + arow * (long)K + g * 16;            // A frag: 16 contiguous K at g*16
  const u16* bp = BT + ((long)(tn4 * 64 + i)) * K + g * 16; // B frag: 16 contiguous K at g*16

  v8f acc0 = {}, acc1 = {}, acc2 = {}, acc3 = {};
#pragma unroll 4
  for (int kb = 0; kb < K; kb += 32) {
    v8bf alo = *(const v8bf*)(ap);
    v8bf ahi = *(const v8bf*)(ap + 8);
    v16bf a = __builtin_shufflevector(alo, ahi,
              0,1,2,3,4,5,6,7,8,9,10,11,12,13,14,15);
    v16bf b0 = *(const v16bf*)(bp);
    v16bf b1 = *(const v16bf*)(bp + 16L * K);
    v16bf b2 = *(const v16bf*)(bp + 32L * K);
    v16bf b3 = *(const v16bf*)(bp + 48L * K);
    acc0 = __builtin_amdgcn_wmma_f32_16x16x32_bf16(false, a, false, b0, (short)0, acc0, false, false);
    acc1 = __builtin_amdgcn_wmma_f32_16x16x32_bf16(false, a, false, b1, (short)0, acc1, false, false);
    acc2 = __builtin_amdgcn_wmma_f32_16x16x32_bf16(false, a, false, b2, (short)0, acc2, false, false);
    acc3 = __builtin_amdgcn_wmma_f32_16x16x32_bf16(false, a, false, b3, (short)0, acc3, false, false);
    ap += 32; bp += 32;
    __builtin_prefetch(ap + 96, 0, 1);
    __builtin_prefetch(bp + 96, 0, 1);
  }
  const int rbase = (int)(tm * 16) + 8 * g;       // C layout: row = r + 8*g, col = i
#pragma unroll
  for (int j = 0; j < 4; ++j) {
    const v8f acc = (j == 0) ? acc0 : (j == 1) ? acc1 : (j == 2) ? acc2 : acc3;
    const int col = tn4 * 64 + j * 16 + i;
    const float bv = bias ? bias[col] : 0.f;
#pragma unroll
    for (int r = 0; r < 8; ++r) {
      long row = rbase + r;
      if (row < M) {
        float val = acc[r] + bv;
        long idx = row * (long)N + col;
        if (out_bf16) ((u16*)C)[idx] = f2bf(val);
        else          ((float*)C)[idx] = val;
      }
    }
  }
}

// ---------------- weight transpose + f32->bf16 (W[K,N] -> WT[N,K]) ----------------
__global__ void transpose_bf16_kernel(const float* __restrict__ W, u16* __restrict__ WT,
                                      int K, int N)
{
  long t = (long)blockIdx.x * blockDim.x + threadIdx.x;
  long total = (long)K * N;
  if (t >= total) return;
  int  n = (int)(t % N);
  long k = t / N;
  WT[(long)n * K + k] = f2bf(W[k * N + n]);
}

// ---------------- fourier noise embedding ----------------
__global__ void fourier_kernel(const float* __restrict__ noise,
                               const float* __restrict__ w0, const float* __restrict__ b0,
                               const float* __restrict__ w1, const float* __restrict__ b1,
                               float* __restrict__ ne, int N)
{
  int n = blockIdx.x * blockDim.x + threadIdx.x;
  if (n >= N) return;
  float x = noise[n];
  const float wconst = 6.28318530718f / 16.0f;
  float s[NUM_FREQ], c[NUM_FREQ];
#pragma unroll
  for (int f = 0; f < NUM_FREQ; ++f) {
    float ph = wconst * x * (float)(f + 1);
    s[f] = __sinf(ph); c[f] = __cosf(ph);
  }
  float h[16];
#pragma unroll
  for (int j = 0; j < 16; ++j) {
    float a = b0[j];
    for (int f = 0; f < NUM_FREQ; ++f) a += s[f] * w0[f * 16 + j];
    for (int f = 0; f < NUM_FREQ; ++f) a += c[f] * w0[(NUM_FREQ + f) * 16 + j];
    h[j] = a / (1.f + __expf(-a));                       // silu
  }
#pragma unroll
  for (int j = 0; j < 16; ++j) {
    float a = b1[j];
    for (int k = 0; k < 16; ++k) a += h[k] * w1[k * 16 + j];
    ne[n * 16 + j] = a / (1.f + __expf(-a));
  }
}

// ---------------- edges MLP layer 1 (K=4, VALU) -> bf16 ----------------
__global__ void edges_l1_kernel(const float* __restrict__ ea, const float* __restrict__ w0,
                                const float* __restrict__ b0, u16* __restrict__ h1, long total)
{
  long t = (long)blockIdx.x * blockDim.x + threadIdx.x;
  if (t >= total) return;
  int  j = (int)(t & 255);
  long e = t >> 8;
  const float* a = ea + e * 4;
  float v = b0[j] + a[0]*w0[j] + a[1]*w0[256+j] + a[2]*w0[512+j] + a[3]*w0[768+j];
  h1[t] = f2bf(fmaxf(v, 0.f));
}

// ---------------- layernorm (with affine) over 256-dim rows -> bf16 ----------------
__global__ void ln_rows_kernel(const float* __restrict__ X, const float* __restrict__ g,
                               const float* __restrict__ b, u16* __restrict__ out, long R)
{
  int lane = threadIdx.x & 31;
  long row = (long)blockIdx.x * (blockDim.x >> 5) + (threadIdx.x >> 5);
  if (row >= R) return;
  const float* xr = X + row * 256 + lane * 8;
  float4 a0 = *(const float4*)xr;
  float4 a1 = *(const float4*)(xr + 4);
  float v[8] = {a0.x,a0.y,a0.z,a0.w,a1.x,a1.y,a1.z,a1.w};
  float s = 0.f;
#pragma unroll
  for (int u = 0; u < 8; ++u) s += v[u];
  float mu = wred32(s) * (1.f/256.f);
  float q = 0.f;
#pragma unroll
  for (int u = 0; u < 8; ++u) { float d = v[u]-mu; q += d*d; }
  float rs = rsqrtf(wred32(q) * (1.f/256.f) + 1e-5f);
  int j0 = lane * 8;
#pragma unroll
  for (int u = 0; u < 8; ++u)
    out[row * 256 + j0 + u] = f2bf((v[u]-mu)*rs * g[j0+u] + b[j0+u]);
}

// ---------------- FiLM conditional layernorm -> bf16 ----------------
__global__ void film_kernel(const float* __restrict__ X, const float* __restrict__ ne,
                            const float* __restrict__ cs_w, const float* __restrict__ cs_b,
                            const float* __restrict__ cb_w, const float* __restrict__ cb_b,
                            u16* __restrict__ xn, int N)
{
  int lane = threadIdx.x & 31;
  long n = (long)blockIdx.x * (blockDim.x >> 5) + (threadIdx.x >> 5);
  if (n >= N) return;
  const float* xr = X + n * 256 + lane * 8;
  float4 a0 = *(const float4*)xr;
  float4 a1 = *(const float4*)(xr + 4);
  float v[8] = {a0.x,a0.y,a0.z,a0.w,a1.x,a1.y,a1.z,a1.w};
  float s = 0.f;
#pragma unroll
  for (int u = 0; u < 8; ++u) s += v[u];
  float mu = wred32(s) * (1.f/256.f);
  float q = 0.f;
#pragma unroll
  for (int u = 0; u < 8; ++u) { float d = v[u]-mu; q += d*d; }
  float rs = rsqrtf(wred32(q) * (1.f/256.f) + 1e-5f);
  float c[16];
  const float* nr = ne + n * 16;
#pragma unroll
  for (int k = 0; k < 16; ++k) c[k] = nr[k];
  int j0 = lane * 8;
#pragma unroll
  for (int u = 0; u < 8; ++u) {
    int j = j0 + u;
    float sc = cs_b[j], sh = cb_b[j];
#pragma unroll
    for (int k = 0; k < 16; ++k) {
      sc += c[k] * cs_w[k*256 + j];
      sh += c[k] * cb_w[k*256 + j];
    }
    xn[n*256 + j] = f2bf((v[u]-mu)*rs * (1.f + sc) + sh);
  }
}

// ---------------- fill ----------------
__global__ void fill_f32_kernel(float* __restrict__ p, float v, long n)
{
  long t = (long)blockIdx.x * blockDim.x + threadIdx.x;
  if (t < n) p[t] = v;
}

// ---------------- attention pass 1: logits + segment max (wave per edge) ----------------
__global__ void logits_kernel(const float* __restrict__ q, const float* __restrict__ k,
                              const u16* __restrict__ e, const int* __restrict__ src,
                              const int* __restrict__ dst, float* __restrict__ logits,
                              float* __restrict__ mb, int E, int hc, int cdim)
{
  int lane = threadIdx.x & 31;
  long eid = (long)blockIdx.x * (blockDim.x >> 5) + (threadIdx.x >> 5);
  if (eid >= E) return;
  int s = src[eid], d = dst[eid];
  int cpl = hc >> 5;                      // cols per lane (8 or 32); 8 lanes span one head
  const float* qd = q + (long)d * hc + lane * cpl;
  const float* ks = k + (long)s * hc + lane * cpl;
  const u16*   ee = e + eid * (long)hc + lane * cpl;
  float acc = 0.f;
  for (int u = 0; u < cpl; u += 4) {
    float4 qv = *(const float4*)(qd + u);
    float4 kv = *(const float4*)(ks + u);
    ushort4 ev = *(const ushort4*)(ee + u);
    acc += qv.x * (kv.x + bf2f(ev.x));
    acc += qv.y * (kv.y + bf2f(ev.y));
    acc += qv.z * (kv.z + bf2f(ev.z));
    acc += qv.w * (kv.w + bf2f(ev.w));
  }
  acc = red8(acc);
  if ((lane & 7) == 0) {
    int h = lane >> 3;
    float lg = acc * rsqrtf((float)cdim);
    logits[eid * 4 + h] = lg;
    atomicMaxF(&mb[(long)d * 4 + h], lg);
  }
}

// ---------------- attention pass 2: exp + segment sum ----------------
__global__ void expsum_kernel(const float* __restrict__ mb, float* __restrict__ logits,
                              float* __restrict__ denom, const int* __restrict__ dst, long total)
{
  long t = (long)blockIdx.x * blockDim.x + threadIdx.x;
  if (t >= total) return;
  long eid = t >> 2; int h = (int)(t & 3);
  int d = dst[eid];
  float w = __expf(logits[t] - mb[(long)d * 4 + h]);
  logits[t] = w;
  atomicAdd(&denom[(long)d * 4 + h], w);
}

// ---------------- attention pass 3: weighted scatter of (v[src]+e) ----------------
__global__ void scatter_kernel(const float* __restrict__ v, const u16* __restrict__ e,
                               const float* __restrict__ wlog, const float* __restrict__ denom,
                               const int* __restrict__ src, const int* __restrict__ dst,
                               float* __restrict__ outb, int E, int hc)
{
  int lane = threadIdx.x & 31;
  long eid = (long)blockIdx.x * (blockDim.x >> 5) + (threadIdx.x >> 5);
  if (eid >= E) return;
  int s = src[eid], d = dst[eid];
  int cpl = hc >> 5;
  int h = lane >> 3;
  float alpha = wlog[eid * 4 + h] / (denom[(long)d * 4 + h] + 1e-16f);
  const float* vs = v + (long)s * hc + lane * cpl;
  const u16*   ee = e + eid * (long)hc + lane * cpl;
  float* od = outb + (long)d * hc + lane * cpl;
  for (int u = 0; u < cpl; u += 4) {
    float4 vv = *(const float4*)(vs + u);
    ushort4 ev = *(const ushort4*)(ee + u);
    atomicAdd(od + u + 0, alpha * (vv.x + bf2f(ev.x)));
    atomicAdd(od + u + 1, alpha * (vv.y + bf2f(ev.y)));
    atomicAdd(od + u + 2, alpha * (vv.z + bf2f(ev.z)));
    atomicAdd(od + u + 3, alpha * (vv.w + bf2f(ev.w)));
  }
}

// ---------------- head combine + beta gate (wave per node) ----------------
__global__ void finalize_kernel(const float* __restrict__ outb, const float* __restrict__ rb,
                                const float* __restrict__ wb, float* __restrict__ xo,
                                int N, int hc, int relu)
{
  int lane = threadIdx.x & 31;
  long n = (long)blockIdx.x * (blockDim.x >> 5) + (threadIdx.x >> 5);
  if (n >= N) return;
  int j0 = lane * 8;
  float o[8], r[8];
  if (hc == 256) {                       // concat layout is already [N, H*C]
    const float* p = outb + n * 256 + j0;
#pragma unroll
    for (int u = 0; u < 8; ++u) o[u] = p[u];
  } else {                               // mean over heads
#pragma unroll
    for (int u = 0; u < 8; ++u) o[u] = 0.f;
    for (int h = 0; h < 4; ++h) {
      const float* p = outb + n * 1024 + h * 256 + j0;
#pragma unroll
      for (int u = 0; u < 8; ++u) o[u] += p[u];
    }
#pragma unroll
    for (int u = 0; u < 8; ++u) o[u] *= 0.25f;
  }
  const float* pr = rb + n * 256 + j0;
#pragma unroll
  for (int u = 0; u < 8; ++u) r[u] = pr[u];
  float part = 0.f;
#pragma unroll
  for (int u = 0; u < 8; ++u) {
    int j = j0 + u;
    part += o[u]*wb[j] + r[u]*wb[256+j] + (o[u]-r[u])*wb[512+j];
  }
  part = wred32(part);
  float beta = 1.f / (1.f + __expf(-part));
#pragma unroll
  for (int u = 0; u < 8; ++u) {
    float val = beta * r[u] + (1.f - beta) * o[u];
    if (relu) val = fmaxf(val, 0.f);
    xo[n * 256 + j0 + u] = val;
  }
}

// =====================================================================================
extern "C" void kernel_launch(void* const* d_in, const int* in_sizes, int n_in,
                              void* d_out, int out_size, void* d_ws, size_t ws_size,
                              hipStream_t stream)
{
  (void)in_sizes; (void)n_in; (void)out_size; (void)ws_size;
  const float* x_in  = (const float*)d_in[0];
  const float* eattr = (const float*)d_in[1];
  const int*   eidx  = (const int*)d_in[2];
  const float* noise = (const float*)d_in[3];
  const int* src = eidx;
  const int* dst = eidx + N_EDGES;

  // params flattened: top-level insertion order, nested dicts in sorted-key (JAX pytree) order
  int pi = 4;
  auto nxt = [&]() { return (const float*)d_in[pi++]; };
  struct Blk { const float *be,*bk,*bq,*bs,*bv,*cb_b,*cb_w,*cs_b,*cs_w,*wb,*we,*wk,*wq,*ws,*wv; };
  Blk B[2];
  for (int b = 0; b < 2; ++b) {
    B[b].be=nxt(); B[b].bk=nxt(); B[b].bq=nxt(); B[b].bs=nxt(); B[b].bv=nxt();
    B[b].cb_b=nxt(); B[b].cb_w=nxt(); B[b].cs_b=nxt(); B[b].cs_w=nxt();
    B[b].wb=nxt(); B[b].we=nxt(); B[b].wk=nxt(); B[b].wq=nxt(); B[b].ws=nxt(); B[b].wv=nxt();
  }
  const float *em_b0=nxt(), *em_b1=nxt(), *em_lnb=nxt(), *em_lng=nxt(), *em_w0=nxt(), *em_w1=nxt();
  const float *fo_b0=nxt(), *fo_b1=nxt(), *fo_w0=nxt(), *fo_w1=nxt();

  // ---------------- workspace carve (~1.45 GB) ----------------
  char* base = (char*)d_ws;
  size_t off = 0;
  auto carve = [&](size_t bytes) -> void* {
    void* p = base + off;
    off += (bytes + 255) & ~(size_t)255;
    return p;
  };
  const int HC[2] = {256, 1024};
  u16* w1T = (u16*)carve((size_t)256*256*2);
  u16 *T_wq[2], *T_wk[2], *T_wv[2], *T_we[2], *T_ws[2];
  for (int b = 0; b < 2; ++b) {
    T_wq[b] = (u16*)carve((size_t)HC[b]*256*2);
    T_wk[b] = (u16*)carve((size_t)HC[b]*256*2);
    T_wv[b] = (u16*)carve((size_t)HC[b]*256*2);
    T_we[b] = (u16*)carve((size_t)HC[b]*256*2);
    T_ws[b] = (u16*)carve((size_t)256*256*2);
  }
  float* ne_buf    = (float*)carve((size_t)N_NODES*16*4);
  u16*   edges_emb = (u16*)  carve((size_t)N_EDGES*256*2);
  u16*   xn        = (u16*)  carve((size_t)N_NODES*256*2);
  float* qb        = (float*)carve((size_t)N_NODES*1024*4);
  float* kb        = (float*)carve((size_t)N_NODES*1024*4);
  float* vb        = (float*)carve((size_t)N_NODES*1024*4);
  float* rb        = (float*)carve((size_t)N_NODES*256*4);
  u16*   e_buf     = (u16*)  carve((size_t)N_EDGES*1024*2);
  float* logitsb   = (float*)carve((size_t)N_EDGES*4*4);
  float* mb        = (float*)carve((size_t)N_NODES*4*4);
  float* denomb    = (float*)carve((size_t)N_NODES*4*4);
  float* outb      = (float*)carve((size_t)N_NODES*1024*4);
  float* xcur      = (float*)carve((size_t)N_NODES*256*4);
  u16*   h1  = (u16*)outb;      // alias: consumed before outb is first used
  float* eln = (float*)e_buf;   // alias: consumed before e_buf is first written

  auto gemm = [&](const u16* A, const u16* BT, const float* bias, void* C, int obf,
                  long M, int N, int K) {
    long tiles = ((M + 15) / 16) * (long)(N / 64);   // 16x64 tile per wave
    long blocks = (tiles + 7) / 8;                   // 8 waves / 256 threads per block
    gemm_bf16_kernel<<<dim3((unsigned)blocks), dim3(256), 0, stream>>>(
        A, BT, bias, C, obf, M, N, K);
  };
  auto transpose = [&](const float* W, u16* WT, int K, int N) {
    long total = (long)K * N;
    transpose_bf16_kernel<<<dim3((unsigned)((total + 255) / 256)), dim3(256), 0, stream>>>(
        W, WT, K, N);
  };
  auto fill = [&](float* p, float v, long n) {
    fill_f32_kernel<<<dim3((unsigned)((n + 255) / 256)), dim3(256), 0, stream>>>(p, v, n);
  };

  // ---------------- weight prep (bf16 transposed for WMMA B fragments) ----------------
  transpose(em_w1, w1T, 256, 256);
  for (int b = 0; b < 2; ++b) {
    transpose(B[b].wq, T_wq[b], 256, HC[b]);
    transpose(B[b].wk, T_wk[b], 256, HC[b]);
    transpose(B[b].wv, T_wv[b], 256, HC[b]);
    transpose(B[b].we, T_we[b], 256, HC[b]);
    transpose(B[b].ws, T_ws[b], 256, 256);
  }

  // ---------------- fourier noise embedding ----------------
  fourier_kernel<<<dim3((N_NODES + 255) / 256), dim3(256), 0, stream>>>(
      noise, fo_w0, fo_b0, fo_w1, fo_b1, ne_buf, N_NODES);

  // ---------------- edges MLP: l1 (VALU) -> WMMA GEMM -> LN -> bf16 ----------------
  {
    long total = (long)N_EDGES * 256;
    edges_l1_kernel<<<dim3((unsigned)((total + 255) / 256)), dim3(256), 0, stream>>>(
        eattr, em_w0, em_b0, h1, total);
    gemm(h1, w1T, em_b1, eln, /*f32 out*/0, N_EDGES, 256, 256);
    ln_rows_kernel<<<dim3((N_EDGES + 7) / 8), dim3(256), 0, stream>>>(
        eln, em_lng, em_lnb, edges_emb, (long)N_EDGES);
  }

  // ---------------- transformer blocks ----------------
  const float* xin = x_in;
  for (int b = 0; b < 2; ++b) {
    const int hc = HC[b];
    const int cdim = hc / HEADS;
    const int relu = (b == 0);

    film_kernel<<<dim3((N_NODES + 7) / 8), dim3(256), 0, stream>>>(
        xin, ne_buf, B[b].cs_w, B[b].cs_b, B[b].cb_w, B[b].cb_b, xn, N_NODES);

    gemm(xn, T_wq[b], B[b].bq, qb, 0, N_NODES, hc, 256);
    gemm(xn, T_wk[b], B[b].bk, kb, 0, N_NODES, hc, 256);
    gemm(xn, T_wv[b], B[b].bv, vb, 0, N_NODES, hc, 256);
    gemm(xn, T_ws[b], B[b].bs, rb, 0, N_NODES, 256, 256);
    gemm(edges_emb, T_we[b], B[b].be, e_buf, /*bf16 out*/1, N_EDGES, hc, 256);

    fill(mb, -1e30f, (long)N_NODES * 4);
    fill(denomb, 0.f, (long)N_NODES * 4);
    fill(outb, 0.f, (long)N_NODES * hc);

    logits_kernel<<<dim3((N_EDGES + 7) / 8), dim3(256), 0, stream>>>(
        qb, kb, e_buf, src, dst, logitsb, mb, N_EDGES, hc, cdim);
    {
      long total = (long)N_EDGES * 4;
      expsum_kernel<<<dim3((unsigned)((total + 255) / 256)), dim3(256), 0, stream>>>(
          mb, logitsb, denomb, dst, total);
    }
    scatter_kernel<<<dim3((N_EDGES + 7) / 8), dim3(256), 0, stream>>>(
        vb, e_buf, logitsb, denomb, src, dst, outb, N_EDGES, hc);

    float* xdst = (b == 0) ? xcur : (float*)d_out;
    finalize_kernel<<<dim3((N_NODES + 7) / 8), dim3(256), 0, stream>>>(
        outb, rb, B[b].wb, xdst, N_NODES, hc, relu);
    xin = xcur;
  }
}
